// LearnedLoss_29360396435808
// MI455X (gfx1250) — matrix operand order repaired and checked by reading, not compiled, not run on hardware
//
#include <hip/hip_runtime.h>
#include <hip/hip_bf16.h>
#include <math.h>

#define B_ROWS 8192
#define D_DIM  2048
#define C_CLS  1000
#define LDC    1008   // 63*16, row stride for logits scratch

typedef __attribute__((ext_vector_type(16))) __bf16 v16bf;
typedef __attribute__((ext_vector_type(8)))  __bf16 v8bf;
typedef __attribute__((ext_vector_type(8)))  float  v8f;

// ---------------- fp32 -> bf16 (round to nearest even) ----------------
__device__ __forceinline__ unsigned short f2bf(float f) {
  union { float f; unsigned int u; } x; x.f = f;
  unsigned int u = x.u;
  unsigned int r = u + 0x7FFFu + ((u >> 16) & 1u);
  return (unsigned short)(r >> 16);
}

__global__ void __launch_bounds__(256)
k_convert(const float* __restrict__ src, unsigned short* __restrict__ dst, int n) {
  int i = (blockIdx.x * 256 + threadIdx.x) * 4;
  if (i + 3 < n) {
    float4 v = *(const float4*)(src + i);
    unsigned long long pack =  (unsigned long long)f2bf(v.x)
                            | ((unsigned long long)f2bf(v.y) << 16)
                            | ((unsigned long long)f2bf(v.z) << 32)
                            | ((unsigned long long)f2bf(v.w) << 48);
    *(unsigned long long*)(dst + i) = pack;
  } else {
    for (; i < n; ++i) dst[i] = f2bf(src[i]);
  }
}

// ---------------- bf16 WMMA GEMM: logits = A @ W^T + bias ----------------
// Wave computes a 32x64 tile: 2 A-fragments (M blocking) x 4 B-fragments (N blocking).
// Each B fragment is reused by 2 WMMAs; each A fragment by 4 -> 12 loads / 8 WMMAs per K-step.
__global__ void __launch_bounds__(256)
k_gemm(const unsigned short* __restrict__ Abf, const unsigned short* __restrict__ Wbf,
       const float* __restrict__ bias, float* __restrict__ logits) {
  const int wave  = (blockIdx.x * 256 + (int)threadIdx.x) >> 5;
  const int lane  = threadIdx.x & 31;
  const int l16   = lane & 15;
  const int hi    = (lane >> 4) & 1;        // lanes 16..31
  const int mTile = wave >> 4;              // 0..255  (8192/32 row tiles)
  const int nSup  = wave & 15;              // 0..15   (16 * 64 = 1024 >= 1000 cols)
  const int mBase = mTile * 32;
  const int nBase = nSup * 64;

  v8f acc[2][4];
#pragma unroll
  for (int m = 0; m < 2; ++m)
#pragma unroll
    for (int j = 0; j < 4; ++j)
#pragma unroll
      for (int r = 0; r < 8; ++r) acc[m][j][r] = 0.0f;

  // A: contiguous k order used here (not the ISA table order): lanes<16 hold K0..15, lanes>=16 K16..31 (matches B)
  const unsigned short* aRow0 = Abf + (size_t)(mBase + l16) * D_DIM;
  const unsigned short* aRow1 = aRow0 + (size_t)16 * D_DIM;
  // B: lane n holds column N=n, lanes<16 K0..15, lanes>=16 K16..31 (K-contiguous in W row)
  const unsigned short* bRow[4];
  int nIdx[4];
#pragma unroll
  for (int j = 0; j < 4; ++j) {
    int n = nBase + j * 16 + l16;
    nIdx[j] = n;
    int nc = n < C_CLS ? n : (C_CLS - 1);   // clamp OOB columns (masked at store)
    bRow[j] = Wbf + (size_t)nc * D_DIM;
  }

  const int aOff = hi ? 16 : 0;
  const int bOff = hi ? 16 : 0;

  for (int k = 0; k < D_DIM; k += 32) {
    v16bf a0, a1;
    {
      const v8bf lo0 = *(const v8bf*)(aRow0 + k + aOff);
      const v8bf hi0 = *(const v8bf*)(aRow0 + k + aOff + 8);
      const v8bf lo1 = *(const v8bf*)(aRow1 + k + aOff);
      const v8bf hi1 = *(const v8bf*)(aRow1 + k + aOff + 8);
#pragma unroll
      for (int i = 0; i < 8; ++i) {
        a0[i] = lo0[i]; a0[i + 8] = hi0[i];
        a1[i] = lo1[i]; a1[i + 8] = hi1[i];
      }
    }
#pragma unroll
    for (int j = 0; j < 4; ++j) {
      const v16bf b = *(const v16bf*)(bRow[j] + k + bOff);
      acc[0][j] = __builtin_amdgcn_wmma_f32_16x16x32_bf16(
          false, a0, false, b, (short)0, acc[0][j], false, false);
      acc[1][j] = __builtin_amdgcn_wmma_f32_16x16x32_bf16(
          false, a1, false, b, (short)0, acc[1][j], false, false);
    }
  }

  // C/D layout: VGPR r -> M = mBase + 16*m + r + 8*hi, N = nBase + j*16 + l16
#pragma unroll
  for (int j = 0; j < 4; ++j) {
    int n = nIdx[j];
    if (n < C_CLS) {
      float bb = bias[n];
#pragma unroll
      for (int m = 0; m < 2; ++m)
#pragma unroll
        for (int r = 0; r < 8; ++r) {
          int row = mBase + 16 * m + r + hi * 8;
          logits[(size_t)row * LDC + n] = acc[m][j][r] + bb;
        }
    }
  }
}

// ---------------- per-row reductions: argmax / log-softmax / hinge / entropy ----------------
__global__ void __launch_bounds__(256)
k_row(const float* __restrict__ logits, const int* __restrict__ targets,
      float* __restrict__ accum) {
  __shared__ float sv[256];
  __shared__ int   si[256];
  __shared__ float s0[256], s1[256], s2[256];

  const int row = blockIdx.x;
  const int t   = threadIdx.x;
  const float* rowp = logits + (size_t)row * LDC;

  float v[4];
#pragma unroll
  for (int j = 0; j < 4; ++j) {
    int c = t + 256 * j;
    v[j] = (c < C_CLS) ? rowp[c] : -3.0e38f;
  }

  // local first-index max (own columns are increasing -> strict >)
  float lmax = v[0]; int lidx = t;
#pragma unroll
  for (int j = 1; j < 4; ++j) {
    int c = t + 256 * j;
    if (v[j] > lmax) { lmax = v[j]; lidx = c; }
  }
  sv[t] = lmax; si[t] = lidx;
  __syncthreads();
  for (int s = 128; s > 0; s >>= 1) {
    if (t < s) {
      float ov = sv[t + s]; int oi = si[t + s];
      if (ov > sv[t] || (ov == sv[t] && oi < si[t])) { sv[t] = ov; si[t] = oi; }
    }
    __syncthreads();
  }
  const float rowmax = sv[0];
  const int   pred   = si[0];
  __syncthreads();

  const int tgt = targets[row];
  const int lab = (tgt < 0) ? pred : tgt;        // pseudo-label fill
  const float xt = rowp[lab];

  float pe = 0.0f, px = 0.0f, ph = 0.0f;
#pragma unroll
  for (int j = 0; j < 4; ++j) {
    int c = t + 256 * j;
    if (c < C_CLS) {
      float x = v[j];
      float e = __expf(x - rowmax);
      pe += e;
      px += x * e;
      ph += fmaxf(1.0f - xt + x, 0.0f);
    }
  }
  s0[t] = pe; s1[t] = px; s2[t] = ph;
  __syncthreads();
  for (int s = 128; s > 0; s >>= 1) {
    if (t < s) { s0[t] += s0[t + s]; s1[t] += s1[t + s]; s2[t] += s2[t + s]; }
    __syncthreads();
  }

  if (t == 0) {
    float sumexp = s0[0];
    float lse    = rowmax + __logf(sumexp);
    float ce     = lse - xt;
    float ent    = lse - s1[0] / sumexp;           // -sum(p*logp)
    float hinge  = (s2[0] - 1.0f) / (float)C_CLS;  // subtract self-term, /C
    float wrong  = (pred != lab) ? 1.0f : 0.0f;
    atomicAdd(&accum[0], ce);
    atomicAdd(&accum[1], hinge);
    atomicAdd(&accum[2], ent);
    atomicAdd(&accum[3], wrong * ent);
  }
}

// ---------------- regularizer reductions (stop-gradient terms) ----------------
__global__ void __launch_bounds__(256)
k_reg(const float* __restrict__ x, const float* __restrict__ x0, int n,
      float* __restrict__ acc4) {
  __shared__ float s[4][256];
  float a0 = 0.f, a1 = 0.f, a2 = 0.f, a3 = 0.f;
  for (int i = blockIdx.x * 256 + threadIdx.x; i < n; i += gridDim.x * 256) {
    float w = x[i], w0 = x0[i], d = w - w0;
    a0 += fabsf(w); a1 += w * w; a2 += fabsf(d); a3 += d * d;
  }
  int t = threadIdx.x;
  s[0][t] = a0; s[1][t] = a1; s[2][t] = a2; s[3][t] = a3;
  __syncthreads();
  for (int st = 128; st > 0; st >>= 1) {
    if (t < st) {
      s[0][t] += s[0][t + st]; s[1][t] += s[1][t + st];
      s[2][t] += s[2][t + st]; s[3][t] += s[3][t + st];
    }
    __syncthreads();
  }
  if (t == 0) {
    atomicAdd(&acc4[0], s[0][0]); atomicAdd(&acc4[1], s[1][0]);
    atomicAdd(&acc4[2], s[2][0]); atomicAdd(&acc4[3], s[3][0]);
  }
}

__global__ void k_init(float* __restrict__ a) {
  if (threadIdx.x < 16) a[threadIdx.x] = 0.0f;
}

__global__ void k_final(const float* __restrict__ accum, const float* __restrict__ hp,
                        float* __restrict__ out) {
  if (threadIdx.x == 0 && blockIdx.x == 0) {
    const float invB  = 1.0f / (float)B_ROWS;
    const float invWN = 1.0f / (float)(C_CLS * D_DIM);
    const float invC  = 1.0f / (float)C_CLS;
    float L[8];
    L[0] = accum[0] * invB;                       // ce
    L[1] = accum[1] * invB;                       // hinge
    L[2] = accum[2] * invB;                       // entropy
    L[3] = accum[3] * invB;                       // dcm
    L[4] = accum[4] * invWN + accum[8]  * invC;   // l1_zero
    L[5] = accum[5] * invWN + accum[9]  * invC;   // l2_zero
    L[6] = accum[6] * invWN + accum[10] * invC;   // l1_init
    L[7] = accum[7] * invWN + accum[11] * invC;   // l2_init
    float r = 0.0f;
#pragma unroll
    for (int i = 0; i < 8; ++i) r += L[i] * hp[i];
    out[0] = r;
  }
}

extern "C" void kernel_launch(void* const* d_in, const int* in_sizes, int n_in,
                              void* d_out, int out_size, void* d_ws, size_t ws_size,
                              hipStream_t stream) {
  (void)in_sizes; (void)n_in; (void)out_size; (void)ws_size;
  const float* inputs  = (const float*)d_in[0];
  const int*   targets = (const int*)  d_in[1];
  const float* W       = (const float*)d_in[2];
  const float* b       = (const float*)d_in[3];
  const float* W0      = (const float*)d_in[4];
  const float* b0      = (const float*)d_in[5];
  const float* hp      = (const float*)d_in[6];

  char* ws = (char*)d_ws;
  const size_t offA = 0;                                   // 8192*2048*2  = 33,554,432 B
  const size_t offW = offA + (size_t)B_ROWS * D_DIM * 2;   // 1000*2048*2  =  4,096,000 B
  const size_t offL = offW + (size_t)C_CLS * D_DIM * 2;    // 8192*1008*4  = 33,030,144 B
  const size_t offS = offL + (size_t)B_ROWS * LDC * 4;     // 16 floats
  unsigned short* Abf    = (unsigned short*)(ws + offA);
  unsigned short* Wbf    = (unsigned short*)(ws + offW);
  float*          logits = (float*)(ws + offL);
  float*          accum  = (float*)(ws + offS);

  k_init<<<1, 32, 0, stream>>>(accum);

  const int nA = B_ROWS * D_DIM;   // 16,777,216
  const int nW = C_CLS * D_DIM;    //  2,048,000
  k_convert<<<(nA / 4 + 255) / 256, 256, 0, stream>>>(inputs, Abf, nA);
  k_convert<<<(nW / 4 + 255) / 256, 256, 0, stream>>>(W, Wbf, nW);

  // 256 row-tiles * 16 col-supertiles = 4096 waves -> 512 blocks of 8 waves
  k_gemm<<<512, 256, 0, stream>>>(Abf, Wbf, b, logits);

  k_row<<<B_ROWS, 256, 0, stream>>>(logits, targets, accum);

  k_reg<<<512, 256, 0, stream>>>(W, W0, nW, accum + 4);
  k_reg<<<4,   256, 0, stream>>>(b, b0, C_CLS, accum + 8);

  k_final<<<1, 1, 0, stream>>>(accum, hp, (float*)d_out);
}
